// Net_84825604096754
// MI455X (gfx1250) — compile-verified, matrix operand order checked by reading, not run on hardware
//
#include <hip/hip_runtime.h>
#include <hip/hip_bf16.h>
#include <math.h>

// ---------------------------------------------------------------------------
// GAT (2-layer, heads=1) + dot-product link decoder for MI455X (gfx1250).
//  - Linear layers via v_wmma_f32_16x16x32_bf16 (wave32, f32 accumulate).
//    A fragments: 2x b128 global loads of pre-converted bf16 activations.
//    B fragments: 2x b128 LDS loads of fragment-swizzled bf16 weights.
//  - Segment softmax / scatter-add via fire-and-forget f32 global atomics
//    (all per-node arrays L2-resident: << 192 MB).
// ---------------------------------------------------------------------------

typedef __attribute__((ext_vector_type(16))) __bf16 v16bf;
typedef __attribute__((ext_vector_type(8)))  float  v8f;
typedef __attribute__((ext_vector_type(2)))  unsigned long long ull2; // 16B/16B-aligned

#define GAT_N_NODES 50000
#define GAT_NEG_SLOPE 0.2f

union AFrag { v16bf v; __bf16 e[16]; ull2 q[2]; };
union CFrag { v8f   v; float  e[8];  };

// ---------------------------------------------------------------- utilities
__global__ void k_fill(float* __restrict__ p, float v, int n) {
  int i = blockIdx.x * blockDim.x + threadIdx.x;
  if (i < n) p[i] = v;
}

__global__ void k_f32_to_bf16(const float* __restrict__ s,
                              __bf16* __restrict__ d, int n) {
  int i = blockIdx.x * blockDim.x + threadIdx.x;
  if (i < n) d[i] = (__bf16)s[i];
}

// Edge (src,dst) with PyG-style appended self loops: e in [0, E+n)
__device__ __forceinline__ void edge_sd(const int* __restrict__ ei, int E,
                                        int e, int& s, int& d) {
  if (e < E) { s = ei[e]; d = ei[E + e]; }
  else       { s = d = e - E; }
}

// ------------------------------------------------------------- WMMA linear
// H[M x N] = X[M x K] * W[K x N]; X pre-converted bf16, f32 accumulate.
// One wave32 computes a 16-row tile (all N columns).
// A-frag layout (ISA 7.12.2, 16-bit A 16x32): lane holds row (lane&15),
//   half i holds K = (lane>=16 ? 8 : 0) + (i&7) + (i>>3)*16.
// B-frag mirrors A with lane = output column; W is swizzled into LDS in
// fragment order so each lane reads its 32-byte fragment contiguously.
template<int K, int N>
__global__ void k_gemm_wmma(const __bf16* __restrict__ X,
                            const float* __restrict__ W,
                            float* __restrict__ H, int M) {
  constexpr int KB = K / 32, NT = N / 16;
  __shared__ __align__(16) __bf16 Ws[KB * NT * 32 * 16];

  // cooperative fragment-swizzled fill: idx = ((kb*NT+nt)*32 + lane)*16 + i
  for (int idx = threadIdx.x; idx < K * N; idx += blockDim.x) {
    int i  = idx & 15;
    int l  = (idx >> 4) & 31;
    int f  = idx >> 9;                 // kb*NT + nt
    int kb = f / NT, nt = f - kb * NT;
    int k  = kb * 32 + ((l & 16) ? 8 : 0) + (i & 7) + ((i >> 3) << 4);
    int n  = nt * 16 + (l & 15);
    Ws[idx] = (__bf16)W[k * N + n];
  }
  __syncthreads();

  const int wave = threadIdx.x >> 5;
  const int lane = threadIdx.x & 31;
  const int tile = blockIdx.x * (blockDim.x >> 5) + wave;
  const int m0 = tile * 16;
  if (m0 >= M) return;                 // M % 16 == 0 here (50000)

  const int halfsel = (lane & 16) ? 8 : 0;
  const int col = lane & 15;

  CFrag acc[NT];
  #pragma unroll
  for (int t = 0; t < NT; ++t)
    #pragma unroll
    for (int r = 0; r < 8; ++r) acc[t].e[r] = 0.f;

  #pragma unroll
  for (int kb = 0; kb < KB; ++kb) {
    AFrag a;                           // 2x global b128, no conversion
    const __bf16* xp = X + (size_t)(m0 + col) * K + kb * 32 + halfsel;
    a.q[0] = *(const ull2*)(xp);       // K = base+0..7
    a.q[1] = *(const ull2*)(xp + 16);  // K = base+16..23

    #pragma unroll
    for (int t = 0; t < NT; ++t) {
      AFrag b;                         // 2x ds b128 from swizzled LDS
      const __bf16* bp = Ws + (((kb * NT + t) * 32 + lane) << 4);
      b.q[0] = *(const ull2*)(bp);
      b.q[1] = *(const ull2*)(bp + 8);
      acc[t].v = __builtin_amdgcn_wmma_f32_16x16x32_bf16(
          false, a.v, false, b.v, (short)0, acc[t].v, false, false);
    }
  }

  // C/D layout: VGPR r -> M = r + 8*(lane>=16), N = lane&15
  const int rbase = m0 + ((lane >> 4) << 3);
  #pragma unroll
  for (int t = 0; t < NT; ++t)
    #pragma unroll
    for (int r = 0; r < 8; ++r)
      H[(size_t)(rbase + r) * N + t * 16 + col] = acc[t].e[r];
}

// -------------------------------------------- per-node attention scalars
template<int F>
__global__ void k_alpha(const float* __restrict__ H,
                        const float* __restrict__ avs,
                        const float* __restrict__ avd,
                        float* __restrict__ s, float* __restrict__ d, int n) {
  int i = blockIdx.x * blockDim.x + threadIdx.x;
  if (i >= n) return;
  const float4* h4 = (const float4*)(H + (size_t)i * F);
  const float4* a4 = (const float4*)avs;
  const float4* b4 = (const float4*)avd;
  float ss = 0.f, dd = 0.f;
  #pragma unroll
  for (int j = 0; j < F / 4; ++j) {
    float4 h = h4[j], a = a4[j], b = b4[j];
    ss += h.x * a.x + h.y * a.y + h.z * a.z + h.w * a.w;
    dd += h.x * b.x + h.y * b.y + h.z * b.z + h.w * b.w;
  }
  s[i] = ss; d[i] = dd;
}

// -------------------------------- segment softmax pass 1: leaky-relu + max
__global__ void k_edge_max(const int* __restrict__ ei, int E, int n,
                           const float* __restrict__ as,
                           const float* __restrict__ ad,
                           float* __restrict__ ebuf, float* __restrict__ mx) {
  int e = blockIdx.x * blockDim.x + threadIdx.x;
  if (e >= E + n) return;
  int s, d; edge_sd(ei, E, e, s, d);
  float v = as[s] + ad[d];
  v = v > 0.f ? v : GAT_NEG_SLOPE * v;
  ebuf[e] = v;
  atomicMax(mx + d, v);                // GLOBAL_ATOMIC_MAX_NUM_F32, no return
}

// -------------------------------- segment softmax pass 2: exp + denom sum
__global__ void k_edge_expsum(const int* __restrict__ ei, int E, int n,
                              const float* __restrict__ mx,
                              float* __restrict__ ebuf,
                              float* __restrict__ dn) {
  int e = blockIdx.x * blockDim.x + threadIdx.x;
  if (e >= E + n) return;
  int s, d; edge_sd(ei, E, e, s, d);
  float ex = __expf(ebuf[e] - mx[d]);
  ebuf[e] = ex;
  atomicAdd(dn + d, ex);               // GLOBAL_ATOMIC_ADD_F32, no return
}

// ------------------------- weighted message scatter: agg[d] += alpha * h[s]
template<int F>
__global__ void k_edge_scatter(const int* __restrict__ ei, int E, int n,
                               const float* __restrict__ ebuf,
                               const float* __restrict__ dn,
                               const float* __restrict__ H,
                               float* __restrict__ agg) {
  const int C = F / 4;
  long long gid = (long long)blockIdx.x * blockDim.x + threadIdx.x;
  long long tot = (long long)(E + n) * C;
  if (gid >= tot) return;
  int e = (int)(gid / C);
  int c = (int)(gid % C);
  int s, d; edge_sd(ei, E, e, s, d);
  float w = ebuf[e] / dn[d];
  float4 h = ((const float4*)(H + (size_t)s * F))[c];
  float* o = agg + (size_t)d * F + c * 4;
  atomicAdd(o + 0, h.x * w);
  atomicAdd(o + 1, h.y * w);
  atomicAdd(o + 2, h.z * w);
  atomicAdd(o + 3, h.w * w);
}

// ------------------- layer-1 epilogue: bias + relu, emit bf16 for GEMM2
template<int F>
__global__ void k_bias_relu_bf16(const float* __restrict__ p,
                                 const float* __restrict__ b,
                                 __bf16* __restrict__ q, int total) {
  int i = blockIdx.x * blockDim.x + threadIdx.x;
  if (i >= total) return;
  float v = p[i] + b[i & (F - 1)];
  q[i] = (__bf16)(v > 0.f ? v : 0.f);
}

// ------------------------------------- layer-2 epilogue: bias only, f32
template<int F>
__global__ void k_bias(float* __restrict__ p, const float* __restrict__ b,
                       int total) {
  int i = blockIdx.x * blockDim.x + threadIdx.x;
  if (i >= total) return;
  p[i] = p[i] + b[i & (F - 1)];
}

// ------------------------------------------- decoder: per-edge z·z (F = 32)
__global__ void k_decode(const int* __restrict__ pos,
                         const int* __restrict__ neg, int ED,
                         const float* __restrict__ Z, float* __restrict__ out) {
  int e = blockIdx.x * blockDim.x + threadIdx.x;
  if (e >= 2 * ED) return;
  int a, b;
  if (e < ED) { a = pos[e];      b = pos[ED + e]; }
  else        { int t = e - ED; a = neg[t]; b = neg[ED + t]; }
  const float4* za = (const float4*)(Z + (size_t)a * 32);
  const float4* zb = (const float4*)(Z + (size_t)b * 32);
  float acc = 0.f;
  #pragma unroll
  for (int j = 0; j < 8; ++j) {
    float4 x = za[j], y = zb[j];
    acc += x.x * y.x + x.y * y.y + x.z * y.z + x.w * y.w;
  }
  out[e] = acc;
}

// ---------------------------------------------------------------------------
extern "C" void kernel_launch(void* const* d_in, const int* in_sizes, int n_in,
                              void* d_out, int out_size, void* d_ws, size_t ws_size,
                              hipStream_t stream) {
  const float* x   = (const float*)d_in[0];
  const int*   ei  = (const int*)  d_in[1];
  const int*   pos = (const int*)  d_in[2];
  const int*   neg = (const int*)  d_in[3];
  const float* W1  = (const float*)d_in[4];
  const float* aS1 = (const float*)d_in[5];
  const float* aD1 = (const float*)d_in[6];
  const float* b1  = (const float*)d_in[7];
  const float* W2  = (const float*)d_in[8];
  const float* aS2 = (const float*)d_in[9];
  const float* aD2 = (const float*)d_in[10];
  const float* b2  = (const float*)d_in[11];
  float* out = (float*)d_out;

  const int n  = GAT_N_NODES;
  const int E  = in_sizes[1] / 2;
  const int ED = in_sizes[2] / 2;
  const int EN = E + n;

  // workspace layout: f32 region then bf16 region (~66 MB total)
  float* ws   = (float*)d_ws;
  float* h1   = ws;                        // n*64  messages layer1
  float* agg1 = h1   + (size_t)n * 64;     // n*64  aggregated layer1
  float* h2   = agg1 + (size_t)n * 64;     // n*32  messages layer2
  float* z    = h2   + (size_t)n * 32;     // n*32  aggregated layer2 (final z)
  float* as1  = z    + (size_t)n * 32;
  float* ad1  = as1 + n;
  float* mx1  = ad1 + n;
  float* dn1  = mx1 + n;
  float* as2  = dn1 + n;
  float* ad2  = as2 + n;
  float* mx2  = ad2 + n;
  float* dn2  = mx2 + n;
  float* e1   = dn2 + n;                   // EN
  float* e2   = e1 + EN;                   // EN
  __bf16* xbf   = (__bf16*)(e2 + EN);      // n*128 bf16 (64B-aligned offset)
  __bf16* agg1b = xbf + (size_t)n * 128;   // n*64  bf16 relu(agg1+b1)

  auto nb = [](long long t, int b) { return (unsigned)((t + b - 1) / b); };
  const int B = 256;

  // ---- init (ws is poisoned; everything read-before-full-write gets filled)
  k_fill<<<nb(n, B), B, 0, stream>>>(mx1, -INFINITY, n);
  k_fill<<<nb(n, B), B, 0, stream>>>(dn1, 0.f, n);
  k_fill<<<nb((long long)n * 64, B), B, 0, stream>>>(agg1, 0.f, n * 64);
  k_fill<<<nb(n, B), B, 0, stream>>>(mx2, -INFINITY, n);
  k_fill<<<nb(n, B), B, 0, stream>>>(dn2, 0.f, n);
  k_fill<<<nb((long long)n * 32, B), B, 0, stream>>>(z, 0.f, n * 32);

  // ---- layer 1: bf16 convert + linear (WMMA) + attention + aggregate + relu
  {
    int tiles = (n + 15) / 16;                       // 3125
    k_f32_to_bf16<<<nb((long long)n * 128, B), B, 0, stream>>>(x, xbf, n * 128);
    k_gemm_wmma<128, 64><<<nb(tiles, 4), 128, 0, stream>>>(xbf, W1, h1, n);
    k_alpha<64><<<nb(n, B), B, 0, stream>>>(h1, aS1, aD1, as1, ad1, n);
    k_edge_max   <<<nb(EN, B), B, 0, stream>>>(ei, E, n, as1, ad1, e1, mx1);
    k_edge_expsum<<<nb(EN, B), B, 0, stream>>>(ei, E, n, mx1, e1, dn1);
    k_edge_scatter<64><<<nb((long long)EN * 16, B), B, 0, stream>>>(
        ei, E, n, e1, dn1, h1, agg1);
    k_bias_relu_bf16<64><<<nb((long long)n * 64, B), B, 0, stream>>>(
        agg1, b1, agg1b, n * 64);
  }

  // ---- layer 2
  {
    int tiles = (n + 15) / 16;
    k_gemm_wmma<64, 32><<<nb(tiles, 4), 128, 0, stream>>>(agg1b, W2, h2, n);
    k_alpha<32><<<nb(n, B), B, 0, stream>>>(h2, aS2, aD2, as2, ad2, n);
    k_edge_max   <<<nb(EN, B), B, 0, stream>>>(ei, E, n, as2, ad2, e2, mx2);
    k_edge_expsum<<<nb(EN, B), B, 0, stream>>>(ei, E, n, mx2, e2, dn2);
    k_edge_scatter<32><<<nb((long long)EN * 8, B), B, 0, stream>>>(
        ei, E, n, e2, dn2, h2, z);
    k_bias<32><<<nb((long long)n * 32, B), B, 0, stream>>>(z, b2, n * 32);
  }

  // ---- decoder
  k_decode<<<nb(2 * ED, B), B, 0, stream>>>(pos, neg, ED, z, out);
}
